// TemporalCrossAttention_19327352832531
// MI455X (gfx1250) — compile-verified, matrix operand order checked by reading, not run on hardware
//
#include <hip/hip_runtime.h>
#include <hip/hip_bf16.h>

typedef __attribute__((ext_vector_type(16))) _Float16 v16h;
typedef __attribute__((ext_vector_type(8)))  _Float16 v8h;
typedef __attribute__((ext_vector_type(8)))  float    v8f;
typedef __attribute__((ext_vector_type(4)))  float    v4f;
typedef __attribute__((ext_vector_type(4)))  int      v4i;

#define B_  4
#define N_  4096
#define M_  4096
#define C_  256
#define D_  256

#if __has_builtin(__builtin_amdgcn_global_load_async_to_lds_b128) && \
    __has_builtin(__builtin_amdgcn_s_wait_asynccnt)
#define HAVE_ASYNC_LDS 1
#else
#define HAVE_ASYNC_LDS 0
#endif

__device__ __forceinline__ v8f wmma_f16(v16h a, v16h b, v8f c) {
  return __builtin_amdgcn_wmma_f32_16x16x32_f16(false, a, false, b, (short)0, c, false, false);
}

// Load 16 f16 in WMMA A/B striped layout: halves [0..7] from p, [8..15] from p+16.
__device__ __forceinline__ v16h load_a16(const _Float16* p) {
  v8h lo = *(const v8h*)(p);
  v8h hi = *(const v8h*)(p + 16);
  v16h r;
#pragma unroll
  for (int i = 0; i < 8; ++i) { r[i] = lo[i]; r[i + 8] = hi[i]; }
  return r;
}

// Same layout but converting from f32 memory.
__device__ __forceinline__ v16h cvt_a16(const float* p) {
  v4f f0 = *(const v4f*)(p);
  v4f f1 = *(const v4f*)(p + 4);
  v4f f2 = *(const v4f*)(p + 16);
  v4f f3 = *(const v4f*)(p + 20);
  v16h r;
#pragma unroll
  for (int i = 0; i < 4; ++i) {
    r[i]      = (_Float16)f0[i];
    r[i + 4]  = (_Float16)f1[i];
    r[i + 8]  = (_Float16)f2[i];
    r[i + 12] = (_Float16)f3[i];
  }
  return r;
}

// 16-byte global -> LDS copy: async (ASYNCcnt-tracked) if available.
__device__ __forceinline__ void cp16(_Float16* ldst, const _Float16* gsrc) {
#if HAVE_ASYNC_LDS
  __builtin_amdgcn_global_load_async_to_lds_b128(
      (__attribute__((address_space(1))) v4i*)gsrc,
      (__attribute__((address_space(3))) v4i*)ldst, 0, 0);
#else
  *(v8h*)ldst = *(const v8h*)gsrc;
#endif
}

#if HAVE_ASYNC_LDS
#define ASYNC_WAIT(n) __builtin_amdgcn_s_wait_asynccnt(n)
#else
#define ASYNC_WAIT(n)
#endif

// ---------------------------------------------------------------------------
// Kernel 1: QKV projection.  One 16x16 tile per wave, K-dim loop of 8 WMMA.
// mat 0: Q = nodes_t  @ Wq^T + bq  -> Qh [b][n][d] f16 (row major)
// mat 1: K = nodes_t1 @ Wk^T + bk  -> Kh [b][m][d] f16 (row major)
// mat 2: V = nodes_t1 @ Wv^T + bv  -> Vt [b][d][m] f16 (transposed for PV B)
// ---------------------------------------------------------------------------
__global__ __launch_bounds__(256) void qkv_proj(
    const float* __restrict__ nt, const float* __restrict__ nt1,
    const float* __restrict__ Wq, const float* __restrict__ bq,
    const float* __restrict__ Wk, const float* __restrict__ bk,
    const float* __restrict__ Wv, const float* __restrict__ bv,
    _Float16* __restrict__ Qh, _Float16* __restrict__ Kh,
    _Float16* __restrict__ Vt)
{
  const int lane = threadIdx.x & 31;
  const int half = lane >> 4;
  const int mn   = lane & 15;
  const int kb   = half * 8;

  int tile = blockIdx.x * 8 + (threadIdx.x >> 5);   // 0..49151
  const int mat  = tile >> 14;                      // 16384 tiles per matrix
  tile &= 16383;
  const int b     = tile >> 12;
  const int rem   = tile & 4095;
  const int rtile = rem >> 4;
  const int dtile = rem & 15;

  const float* X    = (mat == 0) ? nt : nt1;
  const float* W    = (mat == 0) ? Wq : (mat == 1) ? Wk : Wv;
  const float* bias = (mat == 0) ? bq : (mat == 1) ? bk : bv;

  const float* arow = X + (size_t)(b * N_ + rtile * 16 + mn) * C_;
  const float* brow = W + (size_t)(dtile * 16 + mn) * C_;

  v8f acc = {};
#pragma unroll
  for (int c = 0; c < 8; ++c) {
    v16h a  = cvt_a16(arow + c * 32 + kb);
    v16h bm = cvt_a16(brow + c * 32 + kb);
    acc = wmma_f16(a, bm, acc);
  }

  const float bn  = bias[dtile * 16 + mn];
  const int   col = dtile * 16 + mn;
#pragma unroll
  for (int g = 0; g < 8; ++g) {
    const int row = rtile * 16 + g + 8 * half;
    const _Float16 v = (_Float16)(acc[g] + bn);
    if (mat == 0)      Qh[(size_t)(b * N_ + row) * D_ + col] = v;
    else if (mat == 1) Kh[(size_t)(b * M_ + row) * D_ + col] = v;
    else               Vt[(size_t)(b * D_ + col) * M_ + row] = v;
  }
}

// ---------------------------------------------------------------------------
// Kernel 2: flash attention with window mask + online softmax.
// Block = 8 waves, same batch, 128 query rows.  Per 32-key step the block
// cooperatively stages K (32x256) and V^T (256x32) f16 tiles in LDS with
// async global->LDS copies (double buffered, ASYNCcnt), then each wave runs
// 16 WMMA for S and 16 WMMA for PV out of LDS.
// ---------------------------------------------------------------------------
__global__ __launch_bounds__(256) void attn(
    const _Float16* __restrict__ Qh, const _Float16* __restrict__ Kh,
    const _Float16* __restrict__ Vt,
    const float* __restrict__ kpred, const float* __restrict__ kact,
    const int* __restrict__ iterp,
    _Float16* __restrict__ Mh, float* __restrict__ oflow,
    float* __restrict__ oconf)
{
  __shared__ _Float16 Ks[2][32][264];    // K tile, padded rows (33 KB)
  __shared__ _Float16 Vs[2][256][40];    // V^T tile, 80B rows (40 KB)
  __shared__ _Float16 plds[8][16][40];   // per-wave P transpose buffer (10 KB)

  const int lane = threadIdx.x & 31;
  const int half = lane >> 4;
  const int mn   = lane & 15;
  const int kb   = half * 8;
  const int wv   = threadIdx.x >> 5;

  const int wid = blockIdx.x * 8 + wv;   // 0..1023
  const int b   = wid >> 8;              // uniform across the block
  const int n0  = (wid & 255) * 16;

  const int   it  = *iterp;
  const float win = (it == 0) ? 64.0f
                              : fmaxf(4.0f, 32.0f / (float)(1 << (it - 1)));

  // cooperative stager: 256 threads move 16KB K + 16KB V per step (8 x b128/lane)
  auto stage = [&](int j0s, int buf) {
    const int tt   = threadIdx.x;
    const int krow = tt >> 3;
    const int cg   = (tt & 7) * 32;
    const _Float16* ksrc = Kh + (size_t)(b * M_ + j0s + krow) * D_ + cg;
    _Float16*       kdst = &Ks[buf][krow][cg];
#pragma unroll
    for (int i = 0; i < 4; ++i) cp16(kdst + i * 8, ksrc + i * 8);
    const _Float16* vsrc = Vt + (size_t)(b * D_ + tt) * M_ + j0s;
    _Float16*       vdst = &Vs[buf][tt][0];
#pragma unroll
    for (int i = 0; i < 4; ++i) cp16(vdst + i * 8, vsrc + i * 8);
  };

  // Q tile resident in registers (A layout), 8 chunks of K=32.
  v16h qa[8];
  const _Float16* qrow = Qh + (size_t)(b * N_ + n0 + mn) * D_;
#pragma unroll
  for (int c = 0; c < 8; ++c) qa[c] = load_a16(qrow + c * 32 + kb);

  float pu[8], pv[8];
#pragma unroll
  for (int g = 0; g < 8; ++g) {
    const int row = n0 + g + 8 * half;
    pu[g] = kpred[(b * N_ + row) * 2 + 0];
    pv[g] = kpred[(b * N_ + row) * 2 + 1];
  }

  float rm[8], rsum[8], accu[8], accv[8], anyf[8];
#pragma unroll
  for (int g = 0; g < 8; ++g) {
    rm[g] = -1e9f; rsum[g] = 0.f; accu[g] = 0.f; accv[g] = 0.f; anyf[g] = 0.f;
  }
  v8f zero = {};
  v8f o[16];
#pragma unroll
  for (int t = 0; t < 16; ++t) o[t] = zero;

  stage(0, 0);
  for (int step = 0; step < M_ / 32; ++step) {
    const int j0 = step * 32;
    const int cb = step & 1;
    if (step + 1 < M_ / 32) {
      stage(j0 + 32, cb ^ 1);
      ASYNC_WAIT(8);      // current step's 8 copies done; next step's in flight
    } else {
      ASYNC_WAIT(0);
    }
    __syncthreads();

    // ---- S = Q K^T for 32 keys (two 16-col tiles), K from LDS ----
    v8f s0 = zero, s1 = zero;
#pragma unroll
    for (int c = 0; c < 8; ++c) {
      s0 = wmma_f16(qa[c], load_a16(&Ks[cb][mn][c * 32 + kb]), s0);
      s1 = wmma_f16(qa[c], load_a16(&Ks[cb][16 + mn][c * 32 + kb]), s1);
    }

    const int k0 = j0 + mn, k1 = k0 + 16;
    const float au0 = kact[(b * M_ + k0) * 2 + 0];
    const float av0 = kact[(b * M_ + k0) * 2 + 1];
    const float au1 = kact[(b * M_ + k1) * 2 + 0];
    const float av1 = kact[(b * M_ + k1) * 2 + 1];

    // ---- mask + online softmax (row reductions across 16-lane halves) ----
    float p0[8], p1[8], corr[8];
#pragma unroll
    for (int g = 0; g < 8; ++g) {
      float sv0 = s0[g] * 0.0625f;          // 1/sqrt(256)
      float sv1 = s1[g] * 0.0625f;
      const bool m0 = (fabsf(pu[g] - au0) < win) && (fabsf(pv[g] - av0) < win);
      const bool m1 = (fabsf(pu[g] - au1) < win) && (fabsf(pv[g] - av1) < win);
      sv0 = m0 ? sv0 : -1e9f;
      sv1 = m1 ? sv1 : -1e9f;
      anyf[g] = fmaxf(anyf[g], (m0 || m1) ? 1.0f : 0.0f);

      float t = fmaxf(sv0, sv1);
      t = fmaxf(t, __shfl_xor(t, 1, 32));
      t = fmaxf(t, __shfl_xor(t, 2, 32));
      t = fmaxf(t, __shfl_xor(t, 4, 32));
      t = fmaxf(t, __shfl_xor(t, 8, 32));
      const float nm = fmaxf(rm[g], t);
      const float cg = __expf(rm[g] - nm);
      rm[g] = nm;
      p0[g] = __expf(sv0 - nm);
      p1[g] = __expf(sv1 - nm);
      rsum[g] = rsum[g] * cg + p0[g] + p1[g];
      accu[g] = accu[g] * cg + p0[g] * au0 + p1[g] * au1;
      accv[g] = accv[g] * cg + p0[g] * av0 + p1[g] * av1;
      corr[g] = cg;
    }
#pragma unroll
    for (int t = 0; t < 16; ++t)
#pragma unroll
      for (int g = 0; g < 8; ++g) o[t][g] *= corr[g];

    // ---- P: C/D layout -> A layout via per-wave LDS transpose ----
#pragma unroll
    for (int g = 0; g < 8; ++g) {
      plds[wv][g + 8 * half][mn]      = (_Float16)p0[g];
      plds[wv][g + 8 * half][16 + mn] = (_Float16)p1[g];
    }
    v16h pa;
#pragma unroll
    for (int i = 0; i < 8; ++i) {
      pa[i]     = plds[wv][mn][kb + i];
      pa[i + 8] = plds[wv][mn][kb + 16 + i];
    }

    // ---- O += P V : 16 d-tiles, V^T from LDS (contiguous in key) ----
#pragma unroll
    for (int t = 0; t < 16; ++t)
      o[t] = wmma_f16(pa, load_a16(&Vs[cb][t * 16 + mn][kb]), o[t]);

    __syncthreads();
  }

  // ---- epilogue: row reductions, flow/conf, normalized matched features ----
  float invs[8];
#pragma unroll
  for (int g = 0; g < 8; ++g) {
    float rs = rsum[g], au = accu[g], av = accv[g], af = anyf[g];
    rs += __shfl_xor(rs, 1, 32); au += __shfl_xor(au, 1, 32);
    av += __shfl_xor(av, 1, 32); af = fmaxf(af, __shfl_xor(af, 1, 32));
    rs += __shfl_xor(rs, 2, 32); au += __shfl_xor(au, 2, 32);
    av += __shfl_xor(av, 2, 32); af = fmaxf(af, __shfl_xor(af, 2, 32));
    rs += __shfl_xor(rs, 4, 32); au += __shfl_xor(au, 4, 32);
    av += __shfl_xor(av, 4, 32); af = fmaxf(af, __shfl_xor(af, 4, 32));
    rs += __shfl_xor(rs, 8, 32); au += __shfl_xor(au, 8, 32);
    av += __shfl_xor(av, 8, 32); af = fmaxf(af, __shfl_xor(af, 8, 32));
    const float inv = 1.0f / rs;
    invs[g] = inv;
    if (mn == 0) {
      const int row = n0 + g + 8 * half;
      const float conf = af;
      oflow[(b * N_ + row) * 2 + 0] = (au * inv - pu[g]) * conf;
      oflow[(b * N_ + row) * 2 + 1] = (av * inv - pv[g]) * conf;
      oconf[b * N_ + row] = conf;
    }
  }
#pragma unroll
  for (int t = 0; t < 16; ++t)
#pragma unroll
    for (int g = 0; g < 8; ++g)
      Mh[(size_t)(b * N_ + n0 + g + 8 * half) * D_ + t * 16 + mn] =
          (_Float16)(o[t][g] * invs[g]);
}

// ---------------------------------------------------------------------------
// Kernel 3: merged = matched @ Wm^T + bm  (f16 A, converted B, f32 out)
// ---------------------------------------------------------------------------
__global__ __launch_bounds__(256) void merged_gemm(
    const _Float16* __restrict__ Mh, const float* __restrict__ Wm,
    const float* __restrict__ bm, float* __restrict__ out)
{
  const int lane = threadIdx.x & 31;
  const int half = lane >> 4;
  const int mn   = lane & 15;
  const int kb   = half * 8;

  int tile = blockIdx.x * 8 + (threadIdx.x >> 5);  // 0..16383
  const int b     = tile >> 12;
  const int rem   = tile & 4095;
  const int rtile = rem >> 4;
  const int dtile = rem & 15;

  const _Float16* arow = Mh + (size_t)(b * N_ + rtile * 16 + mn) * D_;
  const float*    brow = Wm + (size_t)(dtile * 16 + mn) * D_;

  v8f acc = {};
#pragma unroll
  for (int c = 0; c < 8; ++c)
    acc = wmma_f16(load_a16(arow + c * 32 + kb),
                   cvt_a16(brow + c * 32 + kb), acc);

  const float bn = bm[dtile * 16 + mn];
#pragma unroll
  for (int g = 0; g < 8; ++g)
    out[(size_t)(b * N_ + rtile * 16 + g + 8 * half) * D_ + dtile * 16 + mn] =
        acc[g] + bn;
}

// ---------------------------------------------------------------------------
// Kernel 4: tiny geo MLP (3 -> 32 -> 32), one thread per row.
// ---------------------------------------------------------------------------
__global__ __launch_bounds__(256) void geo_mlp(
    const float* __restrict__ flow, const float* __restrict__ conf,
    const float* __restrict__ Wg1, const float* __restrict__ bg1,
    const float* __restrict__ Wg2, const float* __restrict__ bg2,
    float* __restrict__ geo)
{
  const int i = blockIdx.x * 256 + threadIdx.x;
  if (i >= B_ * N_) return;
  const float fu = flow[i * 2 + 0], fv = flow[i * 2 + 1], cf = conf[i];
  float h[32];
#pragma unroll
  for (int j = 0; j < 32; ++j) {
    const float z = Wg1[j * 3 + 0] * fu + Wg1[j * 3 + 1] * fv +
                    Wg1[j * 3 + 2] * cf + bg1[j];
    h[j] = z / (1.0f + __expf(-z));   // silu
  }
#pragma unroll 4
  for (int j = 0; j < 32; ++j) {
    float z = bg2[j];
#pragma unroll
    for (int k = 0; k < 32; ++k) z += Wg2[j * 32 + k] * h[k];
    geo[(size_t)i * 32 + j] = z;
  }
}

extern "C" void kernel_launch(void* const* d_in, const int* in_sizes, int n_in,
                              void* d_out, int out_size, void* d_ws,
                              size_t ws_size, hipStream_t stream) {
  const float* nodes_t  = (const float*)d_in[0];
  const float* nodes_t1 = (const float*)d_in[1];
  const float* kpred    = (const float*)d_in[2];
  const float* kact     = (const float*)d_in[3];
  const float* Wq  = (const float*)d_in[4];  const float* bq  = (const float*)d_in[5];
  const float* Wk  = (const float*)d_in[6];  const float* bk  = (const float*)d_in[7];
  const float* Wv  = (const float*)d_in[8];  const float* bv  = (const float*)d_in[9];
  const float* Wm  = (const float*)d_in[10]; const float* bm  = (const float*)d_in[11];
  const float* Wg1 = (const float*)d_in[12]; const float* bg1 = (const float*)d_in[13];
  const float* Wg2 = (const float*)d_in[14]; const float* bg2 = (const float*)d_in[15];
  const int*   iterp = (const int*)d_in[16];

  float* out        = (float*)d_out;
  float* out_merged = out;                                  // [4,4096,256]
  float* out_geo    = out_merged + (size_t)B_ * N_ * D_;    // [4,4096,32]
  float* out_flow   = out_geo + (size_t)B_ * N_ * 32;       // [4,4096,2]
  float* out_conf   = out_flow + (size_t)B_ * N_ * 2;       // [4,4096,1]

  _Float16* Qh = (_Float16*)d_ws;                 // 8 MB
  _Float16* Kh = Qh + (size_t)B_ * N_ * D_;       // 8 MB
  _Float16* Vt = Kh + (size_t)B_ * M_ * D_;       // 8 MB (transposed)
  _Float16* Mh = Vt + (size_t)B_ * D_ * M_;       // 8 MB

  qkv_proj<<<6144, 256, 0, stream>>>(nodes_t, nodes_t1, Wq, bq, Wk, bk, Wv, bv,
                                     Qh, Kh, Vt);
  attn<<<128, 256, 0, stream>>>(Qh, Kh, Vt, kpred, kact, iterp, Mh, out_flow,
                                out_conf);
  merged_gemm<<<2048, 256, 0, stream>>>(Mh, Wm, bm, out_merged);
  geo_mlp<<<64, 256, 0, stream>>>(out_flow, out_conf, Wg1, bg1, Wg2, bg2,
                                  out_geo);
}
